// CausalSelfAttention_47940424958039
// MI455X (gfx1250) — compile-verified, matrix operand order checked by reading, not run on hardware
//
#include <hip/hip_runtime.h>
#include <math.h>

// ---------------------------------------------------------------------------
// Fixed problem shape: B=2, T=2048, C=2048, H=16, D=128
// ---------------------------------------------------------------------------
#define Bc 2
#define Tc 2048
#define Cc 2048
#define Hc 16
#define Dc 128

typedef __attribute__((ext_vector_type(16))) __bf16 bf16x16;
typedef __attribute__((ext_vector_type(8)))  __bf16 bf16x8;
typedef __attribute__((ext_vector_type(4)))  __bf16 bf16x4;
typedef __attribute__((ext_vector_type(8)))  float  f32x8;

static __device__ __forceinline__ bf16x16 cat8(bf16x8 lo, bf16x8 hi) {
    return __builtin_shufflevector(lo, hi, 0,1,2,3,4,5,6,7,8,9,10,11,12,13,14,15);
}
static __device__ __forceinline__ bf16x8 ld8(const __bf16* p) {
    return *(const bf16x8*)p;
}
// Fragment: two contiguous 8-elem (16B) runs at contraction offsets h*8 and 16+h*8
// (CDNA5 16-bit A/B operand layout for 16x16x32).
static __device__ __forceinline__ bf16x16 frag32(const __bf16* row, int h) {
    return cat8(ld8(row + h * 8), ld8(row + 16 + h * 8));
}
static __device__ __forceinline__ f32x8 zero8() {
    f32x8 z = {0.f, 0.f, 0.f, 0.f, 0.f, 0.f, 0.f, 0.f};
    return z;
}
static __device__ __forceinline__ f32x8 wmma_bf16(bf16x16 a, bf16x16 b, f32x8 c) {
    return __builtin_amdgcn_wmma_f32_16x16x32_bf16(false, a, false, b, (short)0, c,
                                                   false, false);
}

// ---------------------------------------------------------------------------
// Async global->LDS copy (CDNA5 GLOBAL_LOAD_ASYNC_TO_LDS_B128, ASYNCcnt) with
// a register-staged synchronous fallback if the builtin is absent.
// Builtin parameter 0 is 'int __vector(4) __device__*' per hipcc diagnostic:
// cast to vector_size(16) int with explicit address spaces (AS1 src, AS3 dst).
// ---------------------------------------------------------------------------
#if __has_builtin(__builtin_amdgcn_global_load_async_to_lds_b128)
#define HAS_ASYNC_LDS 1
typedef int v4i __attribute__((vector_size(16)));
typedef __attribute__((address_space(1))) v4i as1_v4i;
typedef __attribute__((address_space(3))) v4i as3_v4i;
#else
#define HAS_ASYNC_LDS 0
#endif

// Copy 32 contiguous bf16 (64B, four b128 chunks) from global to LDS.
static __device__ __forceinline__ void copy64B_g2l(const __bf16* __restrict__ g,
                                                   __bf16* l) {
#if HAS_ASYNC_LDS
#pragma unroll
    for (int i = 0; i < 4; ++i)
        __builtin_amdgcn_global_load_async_to_lds_b128(
            (as1_v4i*)(g + i * 8), (as3_v4i*)(l + i * 8), 0, 0);
#else
    bf16x8 t[4];
#pragma unroll
    for (int i = 0; i < 4; ++i) t[i] = *(const bf16x8*)(g + i * 8);
#pragma unroll
    for (int i = 0; i < 4; ++i) *(bf16x8*)(l + i * 8) = t[i];
#endif
}
static __device__ __forceinline__ void wait_copies() {
#if HAS_ASYNC_LDS
#if __has_builtin(__builtin_amdgcn_s_wait_asynccnt)
    __builtin_amdgcn_s_wait_asynccnt(0);
#else
    asm volatile("s_wait_asynccnt 0" ::: "memory");
#endif
#else
    asm volatile("s_wait_dscnt 0" ::: "memory");
#endif
}

// ---------------------------------------------------------------------------
// Precision staging kernels (one-shot, bandwidth-trivial):
// fp32 -> bf16 copy, and fp32 [K][N] -> bf16 [N][K] tiled transpose.
// ---------------------------------------------------------------------------
__global__ __launch_bounds__(256)
void convert_bf16(const float* __restrict__ in, __bf16* __restrict__ out, int n) {
    const int i = (blockIdx.x * blockDim.x + threadIdx.x) * 8;
    if (i >= n) return;
    const float4 a = *(const float4*)(in + i);
    const float4 b = *(const float4*)(in + i + 4);
    bf16x8 v = {(__bf16)a.x, (__bf16)a.y, (__bf16)a.z, (__bf16)a.w,
                (__bf16)b.x, (__bf16)b.y, (__bf16)b.z, (__bf16)b.w};
    *(bf16x8*)(out + i) = v;
}

__global__ __launch_bounds__(256)
void transpose_bf16(const float* __restrict__ in, __bf16* __restrict__ out,
                    int Kdim, int Ndim) {  // in[K][N] -> out[N][K]
    __shared__ __bf16 tile[64][65];
    const int n0 = blockIdx.x * 64;
    const int k0 = blockIdx.y * 64;
    const int tid = threadIdx.x;
    {
        const int r  = tid >> 2;            // 0..63 (k within tile)
        const int c0 = (tid & 3) * 16;      // 0,16,32,48
#pragma unroll
        for (int i = 0; i < 4; ++i) {
            const float4 v = *(const float4*)(in + (size_t)(k0 + r) * Ndim + n0 + c0 + i * 4);
            tile[r][c0 + i * 4 + 0] = (__bf16)v.x;
            tile[r][c0 + i * 4 + 1] = (__bf16)v.y;
            tile[r][c0 + i * 4 + 2] = (__bf16)v.z;
            tile[r][c0 + i * 4 + 3] = (__bf16)v.w;
        }
    }
    __syncthreads();
    {
        const int rn = tid >> 2;            // 0..63 (n within tile)
        const int c0 = (tid & 3) * 16;      // k within tile
#pragma unroll
        for (int i = 0; i < 4; ++i) {
            bf16x4 p = {tile[c0 + i * 4 + 0][rn], tile[c0 + i * 4 + 1][rn],
                        tile[c0 + i * 4 + 2][rn], tile[c0 + i * 4 + 3][rn]};
            *(bf16x4*)(out + (size_t)(n0 + rn) * Kdim + k0 + c0 + i * 4) = p;
        }
    }
}

// ---------------------------------------------------------------------------
// Tiled bf16-WMMA GEMM:  C[M,N] = A[M,K] * Bt[N,K]^T   (A, Bt bf16 in HBM)
// mode 0: write fp32 C            mode 1: scatter bf16 into head-major Q/K/V
// Block: 256 threads = 8 waves, tile 128x128, K step 64, double-buffered LDS
// fed by async global->LDS copies. Wave (wm,wn) in 2x4 grid owns 64x32.
// ---------------------------------------------------------------------------
__global__ __launch_bounds__(256)
void gemm_wmma_bf16(const __bf16* __restrict__ A, const __bf16* __restrict__ Bt,
                    float* __restrict__ Cout, __bf16* __restrict__ qkv_out,
                    int M, int N, int K, int mode) {
    __shared__ __align__(16) __bf16 As[2][128 * 64];   // [m][k]
    __shared__ __align__(16) __bf16 Bs[2][128 * 64];   // [n][k]

    const int tid  = threadIdx.x;
    const int wave = tid >> 5;
    const int lane = tid & 31;
    const int wm   = wave >> 2;           // 0..1
    const int wn   = wave & 3;            // 0..3
    const int m16  = lane & 15;
    const int h    = lane >> 4;
    const int bm   = blockIdx.y * 128;
    const int bn   = blockIdx.x * 128;

    // per-thread copy slice: row = tid/2 (0..127), 32-elem half-row
    const int crow = tid >> 1;
    const int ccol = (tid & 1) * 32;

    f32x8 acc[4][2];
#pragma unroll
    for (int mt = 0; mt < 4; ++mt)
#pragma unroll
        for (int nt = 0; nt < 2; ++nt) acc[mt][nt] = zero8();

    const int nk = K >> 6;  // K / 64
    // prologue: fill buffer 0
    copy64B_g2l(A + (size_t)(bm + crow) * K + ccol, As[0] + crow * 64 + ccol);
    copy64B_g2l(Bt + (size_t)(bn + crow) * K + ccol, Bs[0] + crow * 64 + ccol);
    wait_copies();
    __syncthreads();

    for (int t = 0; t < nk; ++t) {
        const int cur = t & 1;
        if (t + 1 < nk) {  // issue next tile into the other buffer
            const int k1 = (t + 1) << 6;
            copy64B_g2l(A + (size_t)(bm + crow) * K + k1 + ccol,
                        As[cur ^ 1] + crow * 64 + ccol);
            copy64B_g2l(Bt + (size_t)(bn + crow) * K + k1 + ccol,
                        Bs[cur ^ 1] + crow * 64 + ccol);
        }
        // compute on current buffer: two k=32 sub-steps, 16 WMMAs
#pragma unroll
        for (int kk = 0; kk < 64; kk += 32) {
            bf16x16 af[4];
#pragma unroll
            for (int mt = 0; mt < 4; ++mt)
                af[mt] = frag32(As[cur] + (wm * 64 + mt * 16 + m16) * 64 + kk, h);
            bf16x16 bfr[2];
#pragma unroll
            for (int nt = 0; nt < 2; ++nt)
                bfr[nt] = frag32(Bs[cur] + (wn * 32 + nt * 16 + m16) * 64 + kk, h);
#pragma unroll
            for (int mt = 0; mt < 4; ++mt)
#pragma unroll
                for (int nt = 0; nt < 2; ++nt)
                    acc[mt][nt] = wmma_bf16(af[mt], bfr[nt], acc[mt][nt]);
        }
        wait_copies();
        __syncthreads();
    }

    // ---- epilogue: C element (vgpr r, lane) -> row = r + 8h, col = lane%16
#pragma unroll
    for (int mt = 0; mt < 4; ++mt)
#pragma unroll
        for (int nt = 0; nt < 2; ++nt)
#pragma unroll
            for (int r = 0; r < 8; ++r) {
                const int row = bm + wm * 64 + mt * 16 + r + 8 * h;
                const int col = bn + wn * 32 + nt * 16 + m16;
                const float val = acc[mt][nt][r];
                if (mode == 0) {
                    Cout[(size_t)row * N + col] = val;
                } else {
                    const int which = col >> 11;           // / 2048
                    const int cc    = col & 2047;
                    const int hd    = cc >> 7;             // / 128
                    const int d     = cc & 127;
                    const int b     = row >> 11;
                    const int tt    = row & 2047;
                    qkv_out[((size_t)which * (Bc * Hc) + b * Hc + hd) * (size_t)(Tc * Dc)
                            + (size_t)tt * Dc + d] = (__bf16)val;
                }
            }
}

// ---------------------------------------------------------------------------
// RoPE (GPT-NeoX half-split) applied in place to bf16 [B*H, T, D]
// ---------------------------------------------------------------------------
__global__ __launch_bounds__(256)
void rope_kernel(__bf16* __restrict__ X) {
    const int idx = blockIdx.x * blockDim.x + threadIdx.x;   // BH*T*64 threads
    if (idx >= Bc * Hc * Tc * (Dc / 2)) return;
    const int j  = idx & 63;
    const int t  = (idx >> 6) & (Tc - 1);
    const int bh = idx >> 17;
    __bf16* p = X + ((size_t)bh * Tc + t) * Dc;
    const float x0 = (float)p[j], x1 = (float)p[j + 64];
    const float ang = (float)t * __expf(-0.1439115683121279f * (float)j);
    float s, c;
    __sincosf(ang, &s, &c);
    p[j]      = (__bf16)(x0 * c - x1 * s);
    p[j + 64] = (__bf16)(x1 * c + x0 * s);
}

// ---------------------------------------------------------------------------
// Flash attention, causal.  grid (T/64, B*H), 128 threads = 4 waves.
// Wave w handles q rows [qb0 + 16w, qb0 + 16w + 16).  Key blocks of 32.
// Output written as bf16 [B,T,C] (feeds the out-proj GEMM's bf16 copy path).
// ---------------------------------------------------------------------------
__global__ __launch_bounds__(128)
void attn_wmma(const __bf16* __restrict__ Q, const __bf16* __restrict__ Kv,
               const __bf16* __restrict__ V, __bf16* __restrict__ Y) {
    const int bh  = blockIdx.y;
    const int b   = bh >> 4;
    const int hh  = bh & 15;
    const int qb0 = blockIdx.x * 64;
    const int tid  = threadIdx.x;
    const int wave = tid >> 5;
    const int lane = tid & 31;
    const int n    = lane & 15;
    const int h    = lane >> 4;

    const size_t bh_off = (size_t)bh * Tc * Dc;
    const __bf16* Qb = Q + bh_off;
    const __bf16* Kb = Kv + bh_off;
    const __bf16* Vb = V + bh_off;

    __shared__ __align__(16) __bf16 Vt[Dc * 32];       // [d][key] transposed tile
    __shared__ __align__(16) __bf16 Pl[4][16 * 32];    // per-wave P staging

    // Q fragments for this wave's 16 rows (A operand, row = lane%16)
    bf16x16 qf[4];
    {
        const __bf16* qr = Qb + (size_t)(qb0 + wave * 16 + n) * Dc;
#pragma unroll
        for (int dt = 0; dt < 4; ++dt) qf[dt] = frag32(qr + dt * 32, h);
    }

    f32x8 o[8];
#pragma unroll
    for (int dt = 0; dt < 8; ++dt) o[dt] = zero8();
    float rmax[8], rsum[8];
#pragma unroll
    for (int r = 0; r < 8; ++r) { rmax[r] = -3.0e38f; rsum[r] = 0.f; }

    const float scale = 0.08838834764831845f;  // 1/sqrt(128)
    const int nblk = 2 * blockIdx.x + 2;       // causal bound for this 64-row block

    for (int j = 0; j < nblk; ++j) {
        const int jb = j * 32;
        __syncthreads();
        // ---- cooperative transposed V tile: batch global loads, then LDS stores
        {
            const int key = tid >> 2;            // 0..31
            const int d0  = (tid & 3) * 32;      // 0,32,64,96
            const __bf16* src = Vb + (size_t)(jb + key) * Dc + d0;
            bf16x4 vv[8];
#pragma unroll
            for (int g = 0; g < 8; ++g) vv[g] = *(const bf16x4*)(src + g * 4);
#pragma unroll
            for (int g = 0; g < 8; ++g) {
                Vt[(d0 + g * 4 + 0) * 32 + key] = vv[g][0];
                Vt[(d0 + g * 4 + 1) * 32 + key] = vv[g][1];
                Vt[(d0 + g * 4 + 2) * 32 + key] = vv[g][2];
                Vt[(d0 + g * 4 + 3) * 32 + key] = vv[g][3];
            }
        }
        __syncthreads();

        if (j + 1 < nblk)
            __builtin_prefetch(Vb + (size_t)(jb + 32 + (tid >> 2)) * Dc, 0, 1);

        // ---- S = Q * K^T : two 16x16 tiles over 32 keys, contraction D=128
        f32x8 s[2];
#pragma unroll
        for (int st = 0; st < 2; ++st) {
            const __bf16* kr = Kb + (size_t)(jb + st * 16 + n) * Dc;  // B op: lane = key
            f32x8 a = zero8();
#pragma unroll
            for (int dt = 0; dt < 4; ++dt)
                a = wmma_bf16(qf[dt], frag32(kr + dt * 32, h), a);
            s[st] = a;
        }

        // ---- online softmax (row m = r + 8h, 16 lanes per row)
        float fac[8];
        const int qrow0 = qb0 + wave * 16;
#pragma unroll
        for (int r = 0; r < 8; ++r) {
            const int q  = qrow0 + r + 8 * h;
            const bool m0 = (jb + n)      <= q;
            const bool m1 = (jb + 16 + n) <= q;
            float s0 = m0 ? s[0][r] * scale : -3.0e38f;
            float s1 = m1 ? s[1][r] * scale : -3.0e38f;
            float lm = fmaxf(s0, s1);
#pragma unroll
            for (int off = 8; off >= 1; off >>= 1)
                lm = fmaxf(lm, __shfl_xor(lm, off, 16));
            const float nm = fmaxf(rmax[r], lm);
            const float p0 = m0 ? __expf(s0 - nm) : 0.f;
            const float p1 = m1 ? __expf(s1 - nm) : 0.f;
            float ps = p0 + p1;
#pragma unroll
            for (int off = 8; off >= 1; off >>= 1)
                ps += __shfl_xor(ps, off, 16);
            const float f = __expf(rmax[r] - nm);
            rmax[r] = nm;
            rsum[r] = rsum[r] * f + ps;
            fac[r]  = f;
            __bf16* pl = &Pl[wave][(r + 8 * h) * 32];
            pl[n]      = (__bf16)p0;
            pl[16 + n] = (__bf16)p1;
        }
#pragma unroll
        for (int dt = 0; dt < 8; ++dt)
#pragma unroll
            for (int r = 0; r < 8; ++r) o[dt][r] *= fac[r];

        asm volatile("s_wait_dscnt 0" ::: "memory");   // P stores visible in-wave

        // ---- O += P * V  (P: A op from per-wave LDS; V: B op from Vt)
        const bf16x16 pf = frag32(&Pl[wave][n * 32], h);
#pragma unroll
        for (int dt = 0; dt < 8; ++dt) {
            const bf16x16 vf = frag32(&Vt[(dt * 16 + n) * 32], h);
            o[dt] = wmma_bf16(pf, vf, o[dt]);
        }
    }

    // ---- normalize and write y as bf16 [B, T, C]
#pragma unroll
    for (int r = 0; r < 8; ++r) {
        const float inv = 1.0f / rsum[r];
        const int t = qb0 + wave * 16 + r + 8 * h;
        __bf16* yr = Y + ((size_t)b * Tc + t) * Cc + hh * Dc;
#pragma unroll
        for (int dt = 0; dt < 8; ++dt) yr[dt * 16 + n] = (__bf16)(o[dt][r] * inv);
    }
}

// ---------------------------------------------------------------------------
// Launcher
// ---------------------------------------------------------------------------
extern "C" void kernel_launch(void* const* d_in, const int* in_sizes, int n_in,
                              void* d_out, int out_size, void* d_ws, size_t ws_size,
                              hipStream_t stream) {
    const float* x      = (const float*)d_in[0];   // [B,T,C]
    const float* w_attn = (const float*)d_in[1];   // [C,3C]
    const float* w_proj = (const float*)d_in[2];   // [C,C]
    float* out = (float*)d_out;                    // [B,T,C]

    const size_t bhtd = (size_t)Bc * Hc * Tc * Dc;      // 8388608
    __bf16* Qp  = (__bf16*)d_ws;
    __bf16* Kp  = Qp + bhtd;
    __bf16* Vp  = Qp + 2 * bhtd;
    __bf16* Yb  = Qp + 3 * bhtd;                        // [B,T,C] bf16
    __bf16* Xb  = Qp + 4 * bhtd;                        // [B*T,C] bf16
    __bf16* WaT = Qp + 5 * bhtd;                        // [3C][C] bf16 (transposed)
    __bf16* WpT = WaT + (size_t)3 * Cc * Cc;            // [C][C]  bf16 (transposed)

    const int M  = Bc * Tc;      // 4096
    const int K  = Cc;           // 2048
    const int N1 = 3 * Cc;       // 6144

    // 0) precision staging: x -> bf16, weights -> bf16 transposed [N][K]
    convert_bf16<<<(int)(bhtd / 8 / 256), 256, 0, stream>>>(x, Xb, (int)bhtd);
    transpose_bf16<<<dim3(N1 / 64, K / 64), 256, 0, stream>>>(w_attn, WaT, K, N1);
    transpose_bf16<<<dim3(Cc / 64, K / 64), 256, 0, stream>>>(w_proj, WpT, K, Cc);

    // 1) qkv = x @ w_attn  -> bf16 head-major Q/K/V
    gemm_wmma_bf16<<<dim3(N1 / 128, M / 128), 256, 0, stream>>>(
        Xb, WaT, nullptr, Qp, M, N1, K, 1);

    // 2) RoPE on Q and K
    const int rope_threads = Bc * Hc * Tc * (Dc / 2);
    rope_kernel<<<(rope_threads + 255) / 256, 256, 0, stream>>>(Qp);
    rope_kernel<<<(rope_threads + 255) / 256, 256, 0, stream>>>(Kp);

    // 3) causal flash attention -> y bf16 [B,T,C]
    attn_wmma<<<dim3(Tc / 64, Bc * Hc), 128, 0, stream>>>(Qp, Kp, Vp, Yb);

    // 4) out = y @ w_proj (fp32 out)
    gemm_wmma_bf16<<<dim3(Cc / 128, M / 128), 256, 0, stream>>>(
        Yb, WpT, out, nullptr, M, Cc, K, 0);
}
